// P4TransformerDA7_76905684402578
// MI455X (gfx1250) — matrix operand order repaired and checked by reading, not compiled, not run on hardware
//
#include <hip/hip_runtime.h>
#include <math.h>

// Problem constants (from reference): B=16, N=8192, D=512, M=1024
#define BN_TOK   131072   // B*N tokens
#define DIM      512
#define MSLOTS   1024
#define TILE_R   32       // tokens per workgroup
#define NTHREADS 128      // 4 waves (wave32)
#define INV_SQRT_D 0.044194173824159216f  // 1/sqrt(512)

typedef __attribute__((ext_vector_type(16))) __bf16 v16bf;
typedef __attribute__((ext_vector_type(8)))  float  v8f;

union BF16Frag {
    uint4        q[2];
    unsigned int u[8];
    v16bf        v;
};
static_assert(sizeof(BF16Frag) == 32, "frag must be 32B");

__device__ __forceinline__ unsigned short f2bf(float f) {
    unsigned int u = __float_as_uint(f);
    u += 0x7fffu + ((u >> 16) & 1u);   // round-to-nearest-even
    return (unsigned short)(u >> 16);
}
__device__ __forceinline__ unsigned int pack2bf(float a, float b) {
    return (unsigned int)f2bf(a) | ((unsigned int)f2bf(b) << 16);
}
__device__ __forceinline__ v8f zero8() {
    v8f z;
#pragma unroll
    for (int i = 0; i < 8; ++i) z[i] = 0.0f;
    return z;
}

// ---------------------------------------------------------------------------
// Prep: mem f32 [D][M]  ->  memB bf16 [D][M]  and  memT bf16 [M][D]
// ---------------------------------------------------------------------------
__global__ void memattn_prep_kernel(const float* __restrict__ mem,
                                    unsigned short* __restrict__ memB,
                                    unsigned short* __restrict__ memT) {
    int idx = blockIdx.x * blockDim.x + threadIdx.x;
    if (idx < DIM * MSLOTS) {
        int d = idx / MSLOTS;
        int m = idx - d * MSLOTS;
        unsigned short b = f2bf(mem[idx]);
        memB[idx] = b;                 // [D][M]
        memT[m * DIM + d] = b;         // [M][D]
    }
}

// ---------------------------------------------------------------------------
// Fused: logits = X @ mem / sqrt(D) -> softmax over M -> out = attn @ mem^T
// One workgroup = 32 tokens. 4 waves: wave w -> row stripe (w&1)*16,
// phase-1 M-half / phase-2 D-half = (w>>1).
// ---------------------------------------------------------------------------
__global__ void __launch_bounds__(NTHREADS)
memattn_fused_kernel(const float* __restrict__ x,
                     const unsigned short* __restrict__ memB,
                     const unsigned short* __restrict__ memT,
                     float* __restrict__ out) {
    __shared__ __align__(16) float        Ssm[TILE_R * MSLOTS];      // 128 KB scores/attn (f32)
    __shared__ __align__(16) unsigned int Xs32[TILE_R * (DIM / 2)];  // 32 KB x tile (packed bf16)
    __shared__ float pmax[TILE_R * 4];
    __shared__ float psum[TILE_R * 4];
    __shared__ float rmax[TILE_R];
    __shared__ float rsum[TILE_R];

    const int tid  = threadIdx.x;
    const int wave = tid >> 5;
    const int lane = tid & 31;
    const int half = lane >> 4;        // which 16-lane half
    const int ln   = lane & 15;
    const int t0   = blockIdx.x * TILE_R;

    // ---- Phase 0: load & convert X tile (coalesced f32 -> packed bf16 LDS)
    {
        const float4* xt = (const float4*)(x + (size_t)t0 * DIM);
        for (int i = tid; i < TILE_R * DIM / 4; i += NTHREADS) {
            float4 v = xt[i];
            Xs32[i * 2 + 0] = pack2bf(v.x, v.y);
            Xs32[i * 2 + 1] = pack2bf(v.z, v.w);
        }
    }
    __syncthreads();

    const int rstripe = (wave & 1) * 16;
    const int mhalf   = (wave >> 1) * (MSLOTS / 2);
    const int dhalf   = (wave >> 1) * (DIM / 2);

    // ---- Phase 1: S = X * mem / sqrt(D)   (A = X stripe held in 128 VGPRs)
    {
        BF16Frag Af[16];
#pragma unroll
        for (int s = 0; s < 16; ++s) {
            // A 16x32 bf16 layout: VGPR 0-3: K = k0 + half*8 + 0..7 ; VGPR 4-7: +16
            const unsigned int* p = &Xs32[(rstripe + ln) * (DIM / 2) + s * 16 + half * 4];
            Af[s].q[0] = *(const uint4*)(p);
            Af[s].q[1] = *(const uint4*)(p + 8);
        }

        for (int mt = 0; mt < 32; ++mt) {
            const int m = mhalf + mt * 16 + ln;                 // B column (slot)
            const unsigned short* bp = memT + (size_t)m * DIM + half * 8;
            v8f acc = zero8();
#pragma unroll
            for (int s = 0; s < 16; ++s) {
                BF16Frag Bf;
                const uint4* q = (const uint4*)(bp + s * 32);   // two 16-byte loads, 32 bytes apart / lane
                Bf.q[0] = q[0];
                Bf.q[1] = q[2];
                acc = __builtin_amdgcn_wmma_f32_16x16x32_bf16(
                    false, Af[s].v, false, Bf.v, (short)0, acc, false, false);
            }
#pragma unroll
            for (int j = 0; j < 8; ++j)                          // C: row = j + 8*half
                Ssm[(rstripe + j + half * 8) * MSLOTS + m] = acc[j] * INV_SQRT_D;
        }
    }
    __syncthreads();

    // ---- Softmax over the M axis (exact, f32). 4 threads per token row.
    {
        const int r = tid >> 2;
        const int qq = tid & 3;
        float* row = &Ssm[r * MSLOTS + qq * 256];
        float mx = -3.4e38f;
        for (int i = 0; i < 256; ++i) mx = fmaxf(mx, row[i]);
        pmax[r * 4 + qq] = mx;
        __syncthreads();
        if (qq == 0) {
            float a = fmaxf(fmaxf(pmax[r * 4], pmax[r * 4 + 1]),
                            fmaxf(pmax[r * 4 + 2], pmax[r * 4 + 3]));
            rmax[r] = a;
        }
        __syncthreads();
        const float rm = rmax[r];
        float sm = 0.0f;
        for (int i = 0; i < 256; ++i) {
            float e = __expf(row[i] - rm);
            row[i] = e;
            sm += e;
        }
        psum[r * 4 + qq] = sm;
        __syncthreads();
        if (qq == 0)
            rsum[r] = psum[r * 4] + psum[r * 4 + 1] + psum[r * 4 + 2] + psum[r * 4 + 3];
        __syncthreads();
        const float inv = 1.0f / rsum[r];
        for (int i = 0; i < 256; ++i) row[i] *= inv;
    }
    __syncthreads();

    // ---- Phase 2: out = attn * mem^T   (contract over all M=1024)
    {
        v8f acc2[16];
#pragma unroll
        for (int dt = 0; dt < 16; ++dt) acc2[dt] = zero8();

        for (int chunk = 0; chunk < 4; ++chunk) {
            BF16Frag Af2[8];
#pragma unroll
            for (int kk = 0; kk < 8; ++kk) {
                const int k0 = (chunk * 8 + kk) * 32;
#pragma unroll
                for (int j = 0; j < 8; ++j) {
                    const int g  = j >> 2;
                    const int jj = j & 3;
                    const int kp = k0 + g * 16 + half * 8 + jj * 2;
                    float2 f = *(const float2*)&Ssm[(rstripe + ln) * MSLOTS + kp];
                    Af2[kk].u[j] = pack2bf(f.x, f.y);
                }
            }
#pragma unroll
            for (int dt = 0; dt < 16; ++dt) {
                const int d = dhalf + dt * 16 + ln;             // B column (feature)
                const unsigned short* bp = memB + (size_t)d * MSLOTS + half * 8;
#pragma unroll
                for (int kk = 0; kk < 8; ++kk) {
                    const int k0 = (chunk * 8 + kk) * 32;
                    BF16Frag Bf;
                    const uint4* q = (const uint4*)(bp + k0);
                    Bf.q[0] = q[0];
                    Bf.q[1] = q[2];
                    acc2[dt] = __builtin_amdgcn_wmma_f32_16x16x32_bf16(
                        false, Af2[kk].v, false, Bf.v, (short)0, acc2[dt], false, false);
                }
            }
        }

#pragma unroll
        for (int dt = 0; dt < 16; ++dt) {
            const int col = dhalf + dt * 16 + ln;
#pragma unroll
            for (int j = 0; j < 8; ++j)
                out[(size_t)(t0 + rstripe + j + half * 8) * DIM + col] = acc2[dt][j];
        }
    }
}

// ---------------------------------------------------------------------------
extern "C" void kernel_launch(void* const* d_in, const int* in_sizes, int n_in,
                              void* d_out, int out_size, void* d_ws, size_t ws_size,
                              hipStream_t stream) {
    (void)in_sizes; (void)n_in; (void)out_size; (void)ws_size;

    const float* x   = (const float*)d_in[0];   // [B, N, D] f32
    const float* mem = (const float*)d_in[1];   // [1, D, M] f32
    float* out       = (float*)d_out;           // [B, N, D] f32

    unsigned short* memB = (unsigned short*)d_ws;            // bf16 [D][M], 1 MB
    unsigned short* memT = memB + (size_t)DIM * MSLOTS;      // bf16 [M][D], 1 MB

    memattn_prep_kernel<<<(DIM * MSLOTS + 255) / 256, 256, 0, stream>>>(mem, memB, memT);

    memattn_fused_kernel<<<BN_TOK / TILE_R, NTHREADS, 0, stream>>>(x, memB, memT, out);
}
